// MoEFFN_13013750907088
// MI455X (gfx1250) — compile-verified, matrix operand order checked by reading, not run on hardware
//
#include <hip/hip_runtime.h>
#include <hip/hip_bf16.h>

#define DIM   1024
#define FF    4096
#define NEXP  8
#define NTOK  8192
#define BM    64
#define BN    128
#define BK    32
#define LDSK  40   // element stride for K-dim LDS rows: 40*2B = 80B (16B aligned)

typedef __bf16 bf16_t;
typedef __attribute__((ext_vector_type(2)))  __bf16 v2bf;
typedef __attribute__((ext_vector_type(4)))  __bf16 v4bf;
typedef __attribute__((ext_vector_type(8)))  __bf16 v8bf;
typedef __attribute__((ext_vector_type(16))) __bf16 v16bf;
typedef __attribute__((ext_vector_type(4)))  float  v4f;
typedef __attribute__((ext_vector_type(8)))  float  v8f;

__device__ __forceinline__ v16bf load_frag(const bf16_t* p) {
  // A/B fragment: elems 0..7 = K kb..kb+7, elems 8..15 = K kb+16..kb+23
  v8bf lo = *(const v8bf*)(p);
  v8bf hi = *(const v8bf*)(p + 16);
  return __builtin_shufflevector(lo, hi, 0,1,2,3,4,5,6,7,8,9,10,11,12,13,14,15);
}

__device__ __forceinline__ v4bf pack4(v4f v) {
  v4bf t;
  t[0] = (bf16_t)v[0]; t[1] = (bf16_t)v[1];
  t[2] = (bf16_t)v[2]; t[3] = (bf16_t)v[3];
  return t;
}

__global__ void moe_zero_u32(unsigned int* p, size_t n) {
  size_t i = (size_t)blockIdx.x * blockDim.x + threadIdx.x;
  size_t s = (size_t)gridDim.x * blockDim.x;
  for (; i < n; i += s) p[i] = 0u;
}

__global__ __launch_bounds__(256) void moe_router(
    const float* __restrict__ x, const float* __restrict__ wr,
    int* __restrict__ cnt, int* __restrict__ list, float* __restrict__ glist) {
  int t = blockIdx.x * blockDim.x + threadIdx.x;
  if (t >= NTOK) return;
  const float* xr = x + (size_t)t * DIM;
  float logits[NEXP];
  for (int e = 0; e < NEXP; ++e) {
    const float* w = wr + e * DIM;
    float s = 0.f;
    for (int d = 0; d < DIM; ++d) s = fmaf(xr[d], w[d], s);
    logits[e] = s;
  }
  int i0 = 0; float v0 = logits[0];
  for (int e = 1; e < NEXP; ++e) if (logits[e] > v0) { v0 = logits[e]; i0 = e; }
  int i1 = -1; float v1 = 0.f;
  for (int e = 0; e < NEXP; ++e) {
    if (e == i0) continue;
    if (i1 < 0 || logits[e] > v1) { v1 = logits[e]; i1 = e; }
  }
  float e1 = __expf(v1 - v0);
  float inv = 1.f / (1.f + e1);
  int p0 = atomicAdd(&cnt[i0], 1);
  list[i0 * NTOK + p0]  = t;  glist[i0 * NTOK + p0] = inv;
  int p1 = atomicAdd(&cnt[i1], 1);
  list[i1 * NTOK + p1]  = t;  glist[i1 * NTOK + p1] = e1 * inv;
}

// ---------------- GEMM1: h = silu(Xg @ W1a + b1a) * (Xg @ W1b + b1b) ----------------
__global__ __launch_bounds__(256) void moe_gemm1(
    const float* __restrict__ x, const float* __restrict__ w1, const float* __restrict__ b1,
    const int* __restrict__ cnt_g, const int* __restrict__ list_g,
    bf16_t* __restrict__ h, int e) {
  __shared__ __align__(16) bf16_t Xs[BM * LDSK];
  __shared__ __align__(16) bf16_t WsA[BN * LDSK];
  __shared__ __align__(16) bf16_t WsB[BN * LDSK];
  __shared__ int toks[BM];

  const int cnt = cnt_g[e];
  const int m0 = blockIdx.y * BM;
  if (m0 >= cnt) return;
  const int f0 = blockIdx.x * BN;

  const int tid  = threadIdx.x;
  const int lane = tid & 31;
  const int wv   = tid >> 5;
  const int wm   = (wv >> 2) * 32;  // 0 / 32
  const int wn   = (wv & 3) * 32;   // 0..96
  const int lr   = lane & 15;
  const int kb   = (lane >> 4) * 8;

  if (tid < BM) {
    int r = m0 + tid;
    toks[tid] = (r < cnt) ? list_g[e * NTOK + r] : list_g[e * NTOK];
  }
  __syncthreads();

  // ---- precompute k-invariant staging pointers (straight-line staging) ----
  // X tile: 512 float4 chunks, 2 per thread
  const int xr0 = tid >> 3,           xo0 = (tid & 7) << 2;
  const int xr1 = (tid + 256) >> 3,   xo1 = ((tid + 256) & 7) << 2;
  const float* xs0 = x + (size_t)toks[xr0] * DIM + xo0;
  const float* xs1 = x + (size_t)toks[xr1] * DIM + xo1;
  bf16_t* xd0 = &Xs[xr0 * LDSK + xo0];
  bf16_t* xd1 = &Xs[xr1 * LDSK + xo1];
  // W tiles: chunks cover (k-pair, n4): 512 chunks per half, 2 per thread
  const float* w1e = w1 + (size_t)e * DIM * (2 * FF);
  const int kpA = tid >> 5,          n4A = (tid & 31) << 2;
  const int kpB = (tid + 256) >> 5,  n4B = ((tid + 256) & 31) << 2;
  const float* wsrcA = w1e + (size_t)(2 * kpA) * (2 * FF) + f0 + n4A;
  const float* wsrcB = w1e + (size_t)(2 * kpB) * (2 * FF) + f0 + n4B;

  v8f accA[2][2] = {}; v8f accB[2][2] = {};

  for (int k0 = 0; k0 < DIM; k0 += BK) {
    __syncthreads();
    // X tile: 64x32 f32 -> bf16 (gathered rows), one b64 store per chunk
    *(v4bf*)xd0 = pack4(*(const v4f*)(xs0 + k0));
    *(v4bf*)xd1 = pack4(*(const v4f*)(xs1 + k0));
    // W tiles (both SwiGLU halves), transposed [N][K], packed b32 stores
    {
      const float* s = wsrcA + (size_t)k0 * (2 * FF);
      v4f a0 = *(const v4f*)(s);
      v4f a1 = *(const v4f*)(s + 2 * FF);
      v4f b0 = *(const v4f*)(s + FF);
      v4f b1v = *(const v4f*)(s + FF + 2 * FF);
#pragma unroll
      for (int i = 0; i < 4; ++i) {
        v2bf pa; pa[0] = (bf16_t)a0[i]; pa[1] = (bf16_t)a1[i];
        *(v2bf*)&WsA[(n4A + i) * LDSK + 2 * kpA] = pa;
        v2bf pb; pb[0] = (bf16_t)b0[i]; pb[1] = (bf16_t)b1v[i];
        *(v2bf*)&WsB[(n4A + i) * LDSK + 2 * kpA] = pb;
      }
      if (k0 + BK < DIM) __builtin_prefetch(s + (size_t)BK * (2 * FF), 0, 1);
    }
    {
      const float* s = wsrcB + (size_t)k0 * (2 * FF);
      v4f a0 = *(const v4f*)(s);
      v4f a1 = *(const v4f*)(s + 2 * FF);
      v4f b0 = *(const v4f*)(s + FF);
      v4f b1v = *(const v4f*)(s + FF + 2 * FF);
#pragma unroll
      for (int i = 0; i < 4; ++i) {
        v2bf pa; pa[0] = (bf16_t)a0[i]; pa[1] = (bf16_t)a1[i];
        *(v2bf*)&WsA[(n4B + i) * LDSK + 2 * kpB] = pa;
        v2bf pb; pb[0] = (bf16_t)b0[i]; pb[1] = (bf16_t)b1v[i];
        *(v2bf*)&WsB[(n4B + i) * LDSK + 2 * kpB] = pb;
      }
      if (k0 + BK < DIM) __builtin_prefetch(s + (size_t)BK * (2 * FF), 0, 1);
    }
    __syncthreads();

    v16bf af[2], bfA[2], bfB[2];
#pragma unroll
    for (int i = 0; i < 2; ++i)
      af[i] = load_frag(&Xs[(wm + i * 16 + lr) * LDSK + kb]);
#pragma unroll
    for (int j = 0; j < 2; ++j) {
      bfA[j] = load_frag(&WsA[(wn + j * 16 + lr) * LDSK + kb]);
      bfB[j] = load_frag(&WsB[(wn + j * 16 + lr) * LDSK + kb]);
    }
#pragma unroll
    for (int i = 0; i < 2; ++i)
#pragma unroll
      for (int j = 0; j < 2; ++j) {
        accA[i][j] = __builtin_amdgcn_wmma_f32_16x16x32_bf16(
            false, af[i], false, bfA[j], (short)0, accA[i][j], false, false);
        accB[i][j] = __builtin_amdgcn_wmma_f32_16x16x32_bf16(
            false, af[i], false, bfB[j], (short)0, accB[i][j], false, false);
      }
  }

  const float* b1e = b1 + (size_t)e * (2 * FF);
  const int mrow = 8 * (lane >> 4);
#pragma unroll
  for (int i = 0; i < 2; ++i)
#pragma unroll
    for (int j = 0; j < 2; ++j) {
      int nl = wn + j * 16 + lr;
      float ba = b1e[f0 + nl];
      float bb = b1e[FF + f0 + nl];
#pragma unroll
      for (int r = 0; r < 8; ++r) {
        int ml = wm + i * 16 + r + mrow;
        int mp = m0 + ml;
        if (mp < cnt) {
          float av = accA[i][j][r] + ba;
          float bv = accB[i][j][r] + bb;
          float hv = (av / (1.f + __expf(-av))) * bv;
          h[(size_t)mp * FF + f0 + nl] = (bf16_t)hv;
        }
      }
    }
}

// ---------------- GEMM2: out[token] += gate * (h @ W2 + b2) ----------------
__global__ __launch_bounds__(256) void moe_gemm2(
    const bf16_t* __restrict__ h, const float* __restrict__ w2, const float* __restrict__ b2,
    const int* __restrict__ cnt_g, const int* __restrict__ list_g, const float* __restrict__ glist_g,
    float* __restrict__ out, int e) {
  __shared__ __align__(16) bf16_t Hs[BM * LDSK];
  __shared__ __align__(16) bf16_t Ws[BN * LDSK];
  __shared__ int   toks[BM];
  __shared__ float gs[BM];

  const int cnt = cnt_g[e];
  const int m0 = blockIdx.y * BM;
  if (m0 >= cnt) return;
  const int n0 = blockIdx.x * BN;

  const int tid  = threadIdx.x;
  const int lane = tid & 31;
  const int wv   = tid >> 5;
  const int wm   = (wv >> 2) * 32;
  const int wn   = (wv & 3) * 32;
  const int lr   = lane & 15;
  const int kb   = (lane >> 4) * 8;

  if (tid < BM) {
    int r = m0 + tid;
    bool v = r < cnt;
    toks[tid] = v ? list_g[e * NTOK + r] : 0;
    gs[tid]   = v ? glist_g[e * NTOK + r] : 0.f;
  }
  __syncthreads();

  // ---- H tile staging via CDNA5 async global->LDS DMA (16B per lane) ----
  const int hr = tid >> 2, hc8 = (tid & 3) << 3;
  const int hrr = (m0 + hr < cnt) ? hr : 0;
  const bf16_t* hsrc = h + (size_t)(m0 + hrr) * FF + hc8;
  const unsigned hdst = (unsigned)(size_t)&Hs[hr * LDSK + hc8];
  // W2 tile: (k-pair, n4) chunks, 2 per thread
  const float* w2e = w2 + (size_t)e * FF * DIM;
  const int kpA = tid >> 5,          n4A = (tid & 31) << 2;
  const int kpB = (tid + 256) >> 5,  n4B = ((tid + 256) & 31) << 2;
  const float* wsrcA = w2e + (size_t)(2 * kpA) * DIM + n0 + n4A;
  const float* wsrcB = w2e + (size_t)(2 * kpB) * DIM + n0 + n4B;

  v8f acc[2][2] = {};

  for (int k0 = 0; k0 < FF; k0 += BK) {
    __syncthreads();
    // async copy H tile chunk (already bf16): global -> LDS, ASYNCcnt-tracked
    asm volatile("global_load_async_to_lds_b128 %0, %1, off"
                 :: "v"(hdst), "v"(hsrc + k0) : "memory");
    // W2 tile: 32x128 f32 -> bf16 transposed [N][K], packed b32 stores
    {
      const float* s = wsrcA + (size_t)k0 * DIM;
      v4f a0 = *(const v4f*)(s);
      v4f a1 = *(const v4f*)(s + DIM);
#pragma unroll
      for (int i = 0; i < 4; ++i) {
        v2bf p; p[0] = (bf16_t)a0[i]; p[1] = (bf16_t)a1[i];
        *(v2bf*)&Ws[(n4A + i) * LDSK + 2 * kpA] = p;
      }
      if (k0 + BK < FF) __builtin_prefetch(s + (size_t)BK * DIM, 0, 1);
    }
    {
      const float* s = wsrcB + (size_t)k0 * DIM;
      v4f a0 = *(const v4f*)(s);
      v4f a1 = *(const v4f*)(s + DIM);
#pragma unroll
      for (int i = 0; i < 4; ++i) {
        v2bf p; p[0] = (bf16_t)a0[i]; p[1] = (bf16_t)a1[i];
        *(v2bf*)&Ws[(n4B + i) * LDSK + 2 * kpB] = p;
      }
      if (k0 + BK < FF) __builtin_prefetch(s + (size_t)BK * DIM, 0, 1);
    }
    asm volatile("s_wait_asynccnt 0" ::: "memory");
    __syncthreads();

    v16bf af[2], bfW[2];
#pragma unroll
    for (int i = 0; i < 2; ++i)
      af[i] = load_frag(&Hs[(wm + i * 16 + lr) * LDSK + kb]);
#pragma unroll
    for (int j = 0; j < 2; ++j)
      bfW[j] = load_frag(&Ws[(wn + j * 16 + lr) * LDSK + kb]);
#pragma unroll
    for (int i = 0; i < 2; ++i)
#pragma unroll
      for (int j = 0; j < 2; ++j)
        acc[i][j] = __builtin_amdgcn_wmma_f32_16x16x32_bf16(
            false, af[i], false, bfW[j], (short)0, acc[i][j], false, false);
  }

  const float* b2e = b2 + (size_t)e * DIM;
  const int mrow = 8 * (lane >> 4);
#pragma unroll
  for (int i = 0; i < 2; ++i)
#pragma unroll
    for (int j = 0; j < 2; ++j) {
      int nl = wn + j * 16 + lr;
      float bias = b2e[n0 + nl];
#pragma unroll
      for (int r = 0; r < 8; ++r) {
        int ml = wm + i * 16 + r + mrow;
        int mp = m0 + ml;
        if (mp < cnt) {
          int   tok = toks[ml];
          float g   = gs[ml];
          // experts run sequentially on the stream; one entry per token per expert,
          // so this read-modify-write is race-free and order-deterministic.
          float* o = out + (size_t)tok * DIM + n0 + nl;
          *o = *o + g * (acc[i][j][r] + bias);
        }
      }
    }
}

extern "C" void kernel_launch(void* const* d_in, const int* in_sizes, int n_in,
                              void* d_out, int out_size, void* d_ws, size_t ws_size,
                              hipStream_t stream) {
  const float* x  = (const float*)d_in[0];
  const float* wr = (const float*)d_in[1];
  const float* w1 = (const float*)d_in[2];
  const float* b1 = (const float*)d_in[3];
  const float* w2 = (const float*)d_in[4];
  const float* b2 = (const float*)d_in[5];
  float* out = (float*)d_out;

  char* ws = (char*)d_ws;
  int*    cnt   = (int*)ws;                                      // 8 ints
  int*    list  = (int*)(ws + 256);                              // E*NTOK ints
  float*  glist = (float*)(ws + 256 + NEXP * NTOK * 4);          // E*NTOK f32
  bf16_t* h     = (bf16_t*)(ws + 256 + 2 * NEXP * NTOK * 4);     // NTOK*FF bf16 (reused per expert)

  moe_zero_u32<<<256, 256, 0, stream>>>((unsigned int*)out, (size_t)NTOK * DIM);
  moe_zero_u32<<<1, 32, 0, stream>>>((unsigned int*)cnt, NEXP);
  moe_router<<<NTOK / 256, 256, 0, stream>>>(x, wr, cnt, list, glist);
  for (int e = 0; e < NEXP; ++e) {
    moe_gemm1<<<dim3(FF / BN, NTOK / BM), 256, 0, stream>>>(x, w1, b1, cnt, list, h, e);
    moe_gemm2<<<dim3(DIM / BN, NTOK / BM), 256, 0, stream>>>(h, w2, b2, cnt, list, glist, out, e);
  }
}
